// NF4Linear_81750407512099
// MI455X (gfx1250) — matrix operand order repaired and checked by reading, not compiled, not run on hardware
//
#include <hip/hip_runtime.h>

typedef __attribute__((ext_vector_type(16))) _Float16 v16h;
typedef __attribute__((ext_vector_type(8)))  _Float16 v8h;
typedef __attribute__((ext_vector_type(8)))  float    v8f;
typedef __attribute__((ext_vector_type(4)))  int      v4i;

typedef __attribute__((address_space(1))) v4i* gvec_p;   // global int4*
typedef __attribute__((address_space(3))) v4i* lvec_p;   // LDS int4*

// Problem dims (from reference)
constexpr int M_DIM = 4 * 2048;   // 8192 rows of x
constexpr int N_DIM = 4096;       // out_features
constexpr int K_DIM = 4096;       // in_features
constexpr float DEQ_SCALE = 0.05f / 127.0f;

// GEMM tiling
constexpr int BM = 128;           // block tile M
constexpr int BN = 256;           // block tile N
constexpr int BK = 32;            // block tile K (one WMMA K-step)
constexpr int LDA = 40;           // LDS row pitch (halves), padded 32->40 vs bank conflicts
constexpr int LDB = 40;
constexpr int KTILES = K_DIM / BK;
constexpr int A_TILE = BM * LDA;  // halves per A buffer
constexpr int B_TILE = BN * LDB;  // halves per B buffer

// Workspace layout (bytes): [0, WB) = fp16 W (N x K row-major), then fp16 x (M x K)
constexpr size_t W_HALFS = (size_t)N_DIM * K_DIM;             // 16,777,216
constexpr size_t X_HALFS = (size_t)M_DIM * K_DIM;             // 33,554,432
constexpr size_t W_BYTES = W_HALFS * 2;                       // 33.5 MB

__constant__ float NF4_LUT[16] = {
    -1.0f, -0.6961928009986877f, -0.5250730514526367f, -0.39491748809814453f,
    -0.28444138169288635f, -0.18477343022823334f, -0.09105003625154495f, 0.0f,
    0.07958029955625534f, 0.16093020141124725f, 0.24611230194568634f,
    0.33791524171829224f, 0.44070982933044434f, 0.5626170039176941f,
    0.7229568362236023f, 1.0f};

// ---------------------------------------------------------------------------
// Pass 1: NF4 dequant -> fp16 W.  8 weights per thread (64-block spans 8
// threads, so each thread touches exactly one q_scale).
// ---------------------------------------------------------------------------
__global__ void __launch_bounds__(256) nf4_dequant_kernel(
    const int* __restrict__ idx, const int* __restrict__ qs,
    _Float16* __restrict__ Wh) {
  __shared__ float lut[16];
  if (threadIdx.x < 16) lut[threadIdx.x] = NF4_LUT[threadIdx.x];
  __syncthreads();

  const size_t base = ((size_t)blockIdx.x * 256 + threadIdx.x) * 8;
  const float scale = (float)qs[base >> 6] * DEQ_SCALE;

  const int4* ip = (const int4*)(idx + base);
  const int4 i0 = ip[0];
  const int4 i1 = ip[1];

  v8h h;
  h[0] = (_Float16)(lut[i0.x & 15] * scale);
  h[1] = (_Float16)(lut[i0.y & 15] * scale);
  h[2] = (_Float16)(lut[i0.z & 15] * scale);
  h[3] = (_Float16)(lut[i0.w & 15] * scale);
  h[4] = (_Float16)(lut[i1.x & 15] * scale);
  h[5] = (_Float16)(lut[i1.y & 15] * scale);
  h[6] = (_Float16)(lut[i1.z & 15] * scale);
  h[7] = (_Float16)(lut[i1.w & 15] * scale);
  *(v8h*)(Wh + base) = h;
}

// ---------------------------------------------------------------------------
// Pass 1b: x fp32 -> fp16 (halves GEMM A-side L2 traffic)
// ---------------------------------------------------------------------------
__global__ void __launch_bounds__(256) cvt_x_kernel(
    const float* __restrict__ x, _Float16* __restrict__ Xh) {
  const size_t base = ((size_t)blockIdx.x * 256 + threadIdx.x) * 8;
  const float4* p = (const float4*)(x + base);
  const float4 a = p[0];
  const float4 b = p[1];
  v8h h;
  h[0] = (_Float16)a.x; h[1] = (_Float16)a.y;
  h[2] = (_Float16)a.z; h[3] = (_Float16)a.w;
  h[4] = (_Float16)b.x; h[5] = (_Float16)b.y;
  h[6] = (_Float16)b.z; h[7] = (_Float16)b.w;
  *(v8h*)(Xh + base) = h;
}

// ---------------------------------------------------------------------------
// Pass 2: WMMA GEMM.  y[m,n] = sum_k Xh[m,k] * Wh[n,k]
// 256 threads = 8 waves (2 in M x 4 in N), each wave computes 64x64 via
// 4x4 fragments of v_wmma_f32_16x16x32_f16.  Tiles are pushed into LDS with
// GLOBAL_LOAD_ASYNC_TO_LDS_B128 (ASYNCcnt), double-buffered, one barrier/step.
// K loop is kept un-unrolled (dynamic buffer offset) so the allocator does not
// rotate the 128 accumulator VGPRs into the >256 banked range.
// ---------------------------------------------------------------------------
union Frag16 { v16h v; v8h h[2]; };

__global__ void __launch_bounds__(256) nf4_gemm_kernel(
    const _Float16* __restrict__ Xh, const _Float16* __restrict__ Wh,
    float* __restrict__ out) {
  __shared__ _Float16 lA[2 * A_TILE];   // 2 * 128*40*2B = 20 KB
  __shared__ _Float16 lB[2 * B_TILE];   // 2 * 256*40*2B = 40 KB

  const int tid  = threadIdx.x;
  const int lane = tid & 31;
  const int wave = tid >> 5;
  const int wm   = wave & 1;             // wave M position (0..1)
  const int wn   = wave >> 1;            // wave N position (0..3)
  const int mBase = blockIdx.y * BM;
  const int nBase = blockIdx.x * BN;

  // Cooperative tile-load mapping (256 threads)
  const int aRow = tid >> 1;             // 128 rows of A, 16 halves per thread
  const int aCol = (tid & 1) << 4;
  const int bRow = tid;                  // 256 rows of B(=W), 32 halves per thread

  const _Float16* gA = Xh + (size_t)(mBase + aRow) * K_DIM + aCol;
  const _Float16* gB = Wh + (size_t)(nBase + bRow) * K_DIM;
  _Float16* sA0 = &lA[aRow * LDA + aCol];
  _Float16* sB0 = &lB[bRow * LDB];

  // Async global -> LDS copy of one K-tile (no VGPR staging, no ds_store).
  auto issueTile = [&](int kt, int buf) {
    const _Float16* pa = gA + kt * BK;
    _Float16* sa = sA0 + buf * A_TILE;
    __builtin_amdgcn_global_load_async_to_lds_b128(
        (gvec_p)pa, (lvec_p)sa, 0, 0);
    __builtin_amdgcn_global_load_async_to_lds_b128(
        (gvec_p)(pa + 8), (lvec_p)(sa + 8), 0, 0);
    const _Float16* pb = gB + kt * BK;
    _Float16* sb = sB0 + buf * B_TILE;
#pragma unroll
    for (int j = 0; j < 4; ++j)
      __builtin_amdgcn_global_load_async_to_lds_b128(
          (gvec_p)(pb + 8 * j), (lvec_p)(sb + 8 * j), 0, 0);
  };

  v8f c[4][4] = {};

  const int laneHi = (lane & 16) ? 1 : 0;  // lanes 16..31
  const int lmod   = lane & 15;

  // Per-thread fragment base offsets (halves); mi/ni deltas stay immediates.
  const _Float16* aFrag0 = &lA[(wm * 64 + lmod) * LDA + laneHi * 8];
  const _Float16* bFrag0 = &lB[(wn * 64 + lmod) * LDB + laneHi * 8];

  issueTile(0, 0);

#pragma clang loop unroll(disable)
  for (int kt = 0; kt < KTILES; ++kt) {
    const int buf = kt & 1;
    // Our async copies for tile kt have landed in LDS:
    asm volatile("s_wait_asynccnt 0x0" ::: "memory");
    // ...and everyone else's too:
    __syncthreads();
    // Kick off tile kt+1 into the other buffer; DMA overlaps the WMMAs below.
    // Safe: the barrier above proves all waves finished compute(kt-1), the
    // last reader of buf^1.
    if (kt + 1 < KTILES) issueTile(kt + 1, buf ^ 1);

    // ---- compute on buffer `buf` (dynamic base, immediate sub-offsets) ----
    const _Float16* pA = aFrag0 + buf * A_TILE;
    const _Float16* pB = bFrag0 + buf * B_TILE;

    // A fragments (reused across all 4 B fragments): 32 VGPRs.
    Frag16 af[4];
#pragma unroll
    for (int mi = 0; mi < 4; ++mi) {
      const _Float16* p = pA + mi * (16 * LDA);
      af[mi].h[0] = *(const v8h*)p;         // K = off .. off+7
      af[mi].h[1] = *(const v8h*)(p + 16);  // K = off+16 .. off+23
    }
    // Stream B fragments one at a time (8 VGPRs live instead of 32).
#pragma unroll
    for (int ni = 0; ni < 4; ++ni) {
      Frag16 bf;
      const _Float16* p = pB + ni * (16 * LDB);
      bf.h[0] = *(const v8h*)p;
      bf.h[1] = *(const v8h*)(p + 16);
#pragma unroll
      for (int mi = 0; mi < 4; ++mi)
        c[mi][ni] = __builtin_amdgcn_wmma_f32_16x16x32_f16(
            false, af[mi].v, false, bf.v, (short)0, c[mi][ni], false, false);
    }
  }

  // C/D layout: VGPR v -> M = v (+8 for hi lanes), N = lane&15
#pragma unroll
  for (int mi = 0; mi < 4; ++mi) {
#pragma unroll
    for (int ni = 0; ni < 4; ++ni) {
      const int r0 = mBase + wm * 64 + mi * 16 + laneHi * 8;
      const int cl = nBase + wn * 64 + ni * 16 + lmod;
#pragma unroll
      for (int v = 0; v < 8; ++v)
        out[(size_t)(r0 + v) * N_DIM + cl] = c[mi][ni][v];
    }
  }
}

// ---------------------------------------------------------------------------
extern "C" void kernel_launch(void* const* d_in, const int* in_sizes, int n_in,
                              void* d_out, int out_size, void* d_ws,
                              size_t ws_size, hipStream_t stream) {
  const float* x   = (const float*)d_in[0];
  const int*   idx = (const int*)d_in[1];
  const int*   qs  = (const int*)d_in[2];
  float*       out = (float*)d_out;

  _Float16* Wh = (_Float16*)d_ws;
  _Float16* Xh = (_Float16*)((char*)d_ws + W_BYTES);

  // Pass 1: dequant NF4 -> fp16 W   (16.7M weights, 8 per thread)
  nf4_dequant_kernel<<<(int)(W_HALFS / (256 * 8)), 256, 0, stream>>>(idx, qs, Wh);
  // Pass 1b: x fp32 -> fp16
  cvt_x_kernel<<<(int)(X_HALFS / (256 * 8)), 256, 0, stream>>>(x, Xh);
  // Pass 2: WMMA GEMM
  dim3 grid(N_DIM / BN, M_DIM / BM);    // 16 x 64 workgroups
  nf4_gemm_kernel<<<grid, 256, 0, stream>>>(Xh, Wh, out);
}
